// GraphAttentionLayer_9732395893105
// MI455X (gfx1250) — matrix operand order repaired and checked by reading, not compiled, not run on hardware
//
#include <hip/hip_runtime.h>
#include <hip/hip_bf16.h>

#define BB 4
#define NN 2048
#define DIMN 256
#define HH 4

typedef __attribute__((ext_vector_type(16))) _Float16 v16h;
typedef __attribute__((ext_vector_type(8)))  _Float16 v8h;
typedef __attribute__((ext_vector_type(8)))  float    v8f;
typedef __attribute__((ext_vector_type(4)))  unsigned int u32x4;
typedef __attribute__((ext_vector_type(8)))  int      i32x8;
typedef __attribute__((ext_vector_type(4)))  int      i32x4;

__device__ __forceinline__ void cvt4(v16h& v, int base, float4 f) {
    v[base+0] = (_Float16)f.x; v[base+1] = (_Float16)f.y;
    v[base+2] = (_Float16)f.z; v[base+3] = (_Float16)f.w;
}

// ---------------------------------------------------------------------------
// TDM: DMA a 16-row x 32-col f32 tile (row stride NN floats) into LDS.
// D# packing per CDNA5 ISA 8.3/8.4: group0 = count/lds_addr/global_addr/type,
// group1 = data_size + tensor dims + tile dims + dim0 stride. 2D -> rest = 0.
// Toolchain here is the 6-arg builtin (clang-23 / therock-10.0).
// ---------------------------------------------------------------------------
__device__ __forceinline__ void tdm_load_adj_tile(const float* gsrc, void* ldst) {
    unsigned long long ga = (unsigned long long)(uintptr_t)gsrc;
    unsigned int laddr = (unsigned int)(uintptr_t)ldst;   // LDS offset = addr[31:0]
    u32x4 g0;
    g0[0] = 1u;                                            // count=1 (valid), user mode
    g0[1] = laddr;                                         // lds_addr [63:32]
    g0[2] = (unsigned int)(ga & 0xFFFFFFFFu);              // global_addr lo
    g0[3] = (unsigned int)((ga >> 32) & 0x1FFFFFFu)        // global_addr hi (57-bit)
          | (2u << 30);                                    // type=2 ("image")
    i32x8 g1;
    g1[0] = 0x00020000;            // wg_mask=0, data_size=2 (4 bytes)
    g1[1] = (int)(2048u << 16);    // tensor_dim0 = 2048 (low16 at [31:16])
    g1[2] = (int)(2048u << 16);    // dim0 hi16 = 0 | tensor_dim1 = 2048 (low16)
    g1[3] = (int)(32u << 16);      // dim1 hi16 = 0 | tile_dim0 = 32
    g1[4] = 16;                    // tile_dim1 = 16, tile_dim2 = 0
    g1[5] = 2048;                  // tensor_dim0_stride low32 = 2048 elements
    g1[6] = 0;                     // stride hi16 | tensor_dim1_stride low16
    g1[7] = 0;
    i32x4 z4 = {0, 0, 0, 0};
    i32x8 z8 = {0, 0, 0, 0, 0, 0, 0, 0};
    __builtin_amdgcn_tensor_load_to_lds(g0, g1, z4, z4, z8, 0);
}

// ---------------------------------------------------------------------------
// K0: va1[h,d] = sum_e a[h,e]*W[h,e,d]; va2 uses a[h,DIM+e]; c = a·bias
// ---------------------------------------------------------------------------
__global__ __launch_bounds__(1024) void k_prep(
    const float* __restrict__ W, const float* __restrict__ bias,
    const float* __restrict__ a,
    float* __restrict__ va1, float* __restrict__ va2,
    float* __restrict__ c1, float* __restrict__ c2)
{
    int t = threadIdx.x;          // 1024 = H*DIM
    int h = t >> 8, d = t & 255;
    float s1 = 0.f, s2 = 0.f;
    for (int e = 0; e < DIMN; ++e) {
        float w = W[((size_t)h*DIMN + e)*DIMN + d];
        s1 += a[h*2*DIMN + e]        * w;
        s2 += a[h*2*DIMN + DIMN + e] * w;
    }
    va1[h*DIMN + d] = s1;
    va2[h*DIMN + d] = s2;
    if (t < HH) {
        float t1 = 0.f, t2 = 0.f;
        for (int e = 0; e < DIMN; ++e) {
            t1 += a[t*2*DIMN + e]        * bias[t*DIMN + e];
            t2 += a[t*2*DIMN + DIMN + e] * bias[t*DIMN + e];
        }
        c1[t] = t1; c2[t] = t2;
    }
}

// ---------------------------------------------------------------------------
// K1: embedT[b,h,e,n] (f16) = sum_d x[b,n,d]*W[h,e,d] + bias[h,e]
// WMMA f16 16x16x32; wave covers 32 e-columns, block covers 16 n-rows.
// Stored transposed (e-major) so K4's B fragments are contiguous.
// ---------------------------------------------------------------------------
__global__ __launch_bounds__(256) void k_embed(
    const float* __restrict__ x, const float* __restrict__ W,
    const float* __restrict__ bias, _Float16* __restrict__ embedT)
{
    int it = blockIdx.x & 127;
    int h  = (blockIdx.x >> 7) & 3;
    int b  = blockIdx.x >> 9;
    int i0 = it << 4;
    int lane = threadIdx.x & 31;
    int wave = threadIdx.x >> 5;
    int lc   = lane & 15;           // A-row / B-col index within tile
    int kbA  = (lane >> 4) << 4;    // A K-group base: 0 or 16
    int kbB  = (lane >> 4) << 4;    // B K-group base: 0 or 16
    int e_base = wave << 5;

    const float* xrow = x + ((size_t)b*NN + i0 + lc) * DIMN;
    v8f acc[2] = {};

    for (int ks = 0; ks < DIMN/32; ++ks) {
        int d0 = ks << 5;
        v16h aF;
        {
            const float4* p0 = (const float4*)(xrow + d0 + kbA);
            const float4* p1 = (const float4*)(xrow + d0 + kbA + 8);
            cvt4(aF, 0,  p0[0]); cvt4(aF, 4,  p0[1]);
            cvt4(aF, 8,  p1[0]); cvt4(aF, 12, p1[1]);
        }
        #pragma unroll
        for (int ct = 0; ct < 2; ++ct) {
            const float* wrow = W + ((size_t)h*DIMN + (e_base + ct*16 + lc)) * DIMN;
            const float4* q = (const float4*)(wrow + d0 + kbB);
            v16h bF;
            cvt4(bF, 0, q[0]); cvt4(bF, 4, q[1]);
            cvt4(bF, 8, q[2]); cvt4(bF, 12, q[3]);
            acc[ct] = __builtin_amdgcn_wmma_f32_16x16x32_f16(
                false, aF, false, bF, (short)0, acc[ct], false, false);
        }
    }
    #pragma unroll
    for (int ct = 0; ct < 2; ++ct) {
        int e = e_base + ct*16 + lc;
        float bv = bias[h*DIMN + e];
        v8h outp;
        #pragma unroll
        for (int r = 0; r < 8; ++r) outp[r] = (_Float16)(acc[ct][r] + bv);
        _Float16* dst = embedT + (((size_t)(b*HH + h))*DIMN + e)*NN
                               + i0 + ((lane >> 4) << 3);
        *(v8h*)dst = outp;                      // 16B store
    }
}

// ---------------------------------------------------------------------------
// K2: s_j = x·va1 + c1 ; s_i = x·va2 + c2   (one thread per (b,h,n))
// ---------------------------------------------------------------------------
__global__ __launch_bounds__(256) void k_sij(
    const float* __restrict__ x,
    const float* __restrict__ va1, const float* __restrict__ va2,
    const float* __restrict__ c1,  const float* __restrict__ c2,
    float* __restrict__ s_i, float* __restrict__ s_j)
{
    int gid = blockIdx.x * 256 + threadIdx.x;     // b*H*N + h*N + n
    int n = gid & (NN-1);
    int h = (gid >> 11) & 3;
    int b = gid >> 13;
    const float4* xv = (const float4*)(x + ((size_t)b*NN + n)*DIMN);
    const float4* p1 = (const float4*)(va1 + h*DIMN);
    const float4* p2 = (const float4*)(va2 + h*DIMN);
    float d1 = 0.f, d2 = 0.f;
    for (int k = 0; k < DIMN/4; ++k) {
        float4 xx = xv[k], u = p1[k], w = p2[k];
        d1 += xx.x*u.x + xx.y*u.y + xx.z*u.z + xx.w*u.w;
        d2 += xx.x*w.x + xx.y*w.y + xx.z*w.z + xx.w*w.w;
    }
    s_j[gid] = d1 + c1[h];
    s_i[gid] = d2 + c2[h];
}

// ---------------------------------------------------------------------------
// K3: online softmax stats per (b,h,i): M = rowmax, Z = sum exp(v-M)
// v = lrelu((s_i+s_j)*adj). One wave32 per (b,i), all 4 heads in one pass.
// ---------------------------------------------------------------------------
__global__ __launch_bounds__(256) void k_stats(
    const float* __restrict__ adj,
    const float* __restrict__ s_i, const float* __restrict__ s_j,
    float* __restrict__ Mrow, float* __restrict__ Zrow)
{
    int row  = blockIdx.x * 8 + (threadIdx.x >> 5);   // 0..B*N-1
    int lane = threadIdx.x & 31;
    int b = row >> 11, i = row & (NN-1);
    float si[HH], m[HH], z[HH];
    #pragma unroll
    for (int h = 0; h < HH; ++h) {
        si[h] = s_i[(b*HH + h)*NN + i];
        m[h] = -1e30f; z[h] = 0.f;
    }
    const float* arow = adj + ((size_t)b*NN + i)*NN;
    for (int jt = 0; jt < NN/32; ++jt) {
        int j = jt*32 + lane;
        float av = arow[j];
        #pragma unroll
        for (int h = 0; h < HH; ++h) {
            float v = (si[h] + s_j[(b*HH + h)*NN + j]) * av;
            v = v > 0.f ? v : 0.01f*v;
            float nm = fmaxf(m[h], v);
            z[h] = z[h]*__expf(m[h]-nm) + __expf(v-nm);
            m[h] = nm;
        }
    }
    #pragma unroll
    for (int h = 0; h < HH; ++h) {
        for (int off = 16; off > 0; off >>= 1) {
            float om = __shfl_xor(m[h], off, 32);
            float oz = __shfl_xor(z[h], off, 32);
            float nm = fmaxf(m[h], om);
            z[h] = z[h]*__expf(m[h]-nm) + oz*__expf(om-nm);
            m[h] = nm;
        }
        if (lane == 0) {
            Mrow[(b*HH + h)*NN + i] = m[h];
            Zrow[(b*HH + h)*NN + i] = z[h];
        }
    }
}

// ---------------------------------------------------------------------------
// K4: out[b,i,:] = relu( (1/H) * sum_h attn_h @ embed_h )
// Block = (b, 16-row tile), 8 waves x 32 e-cols = 256 cols.
// j outer (tiles of 32 = WMMA K), h inner: one TDM adj-tile DMA serves all
// 4 heads. Double-buffered TDM prefetch overlaps DMA with WMMA; tileA (f16,
// scaled by 1/Z) is rebuilt per head and consumed as the WMMA A fragment.
// Head-mean accumulates directly in the f32 C registers.
// ---------------------------------------------------------------------------
__global__ __launch_bounds__(256) void k_attn_gemm(
    const float* __restrict__ adj,
    const float* __restrict__ s_i, const float* __restrict__ s_j,
    const float* __restrict__ Mrow, const float* __restrict__ Zrow,
    const _Float16* __restrict__ embedT, float* __restrict__ out)
{
    __shared__ __align__(16) float    adjT[2][16*32];   // TDM destination (double buffer)
    __shared__ __align__(16) _Float16 tileA[16*32];     // attn A-tile (f16)
    int it = blockIdx.x & 127;
    int b  = blockIdx.x >> 7;
    int i0 = it << 4;
    int tid  = threadIdx.x;
    int lane = tid & 31;
    int wave = tid >> 5;
    int lc   = lane & 15;
    int kbA  = (lane >> 4) << 4;    // 0 or 16
    int kbB  = (lane >> 4) << 4;    // 0 or 16
    int i_t = tid >> 4;             // builder row 0..15
    int jl  = (tid & 15) << 1;      // builder col pair

    float si[HH], Mi[HH], iZ[HH];
    #pragma unroll
    for (int h = 0; h < HH; ++h) {
        int rbase = (b*HH + h)*NN;
        si[h] = s_i[rbase + i0 + i_t];
        Mi[h] = Mrow[rbase + i0 + i_t];
        iZ[h] = 1.0f / Zrow[rbase + i0 + i_t];
    }
    const float* atile0 = adj + ((size_t)b*NN + i0)*NN;  // +j0 per step; rows stride NN

    v8f acc0 = {}, acc1 = {};

    if (tid == 0) tdm_load_adj_tile(atile0, &adjT[0][0]);

    for (int js = 0; js < NN/32; ++js) {
        int cur = js & 1;
        int j0  = js << 5;
        if (tid == 0) __builtin_amdgcn_s_wait_tensorcnt(0);   // adjT[cur] landed
        __syncthreads();
        if (tid == 0 && js + 1 < NN/32)                       // prefetch next tile
            tdm_load_adj_tile(atile0 + (js + 1)*32, &adjT[1 - cur][0]);

        float2 a2 = *(const float2*)&adjT[cur][i_t*32 + jl];  // ds read, head-invariant

        #pragma unroll
        for (int h = 0; h < HH; ++h) {
            int rbase = (b*HH + h)*NN;
            // --- build attn tile for head h (one exp per element across WG) ---
            float2 sj2 = *(const float2*)(s_j + rbase + j0 + jl);
            float v0 = (si[h] + sj2.x)*a2.x; v0 = v0 > 0.f ? v0 : 0.01f*v0;
            float v1 = (si[h] + sj2.y)*a2.y; v1 = v1 > 0.f ? v1 : 0.01f*v1;
            tileA[i_t*32 + jl]     = (_Float16)(__expf(v0 - Mi[h])*iZ[h]);
            tileA[i_t*32 + jl + 1] = (_Float16)(__expf(v1 - Mi[h])*iZ[h]);
            __syncthreads();
            // --- A fragment from LDS ---
            v8h alo = *(const v8h*)&tileA[lc*32 + kbA];
            v8h ahi = *(const v8h*)&tileA[lc*32 + kbA + 8];
            v16h aF = __builtin_shufflevector(alo, ahi,
                0,1,2,3,4,5,6,7,8,9,10,11,12,13,14,15);
            // --- B fragments: B(K=j, N=e) = embedT[e][j], K contiguous ---
            const _Float16* eb0 = embedT
                + (((size_t)(b*HH + h))*DIMN + (wave<<5) + lc)*NN;
            const _Float16* eb1 = eb0 + (size_t)16*NN;
            v8h b0lo = *(const v8h*)(eb0 + j0 + kbB);
            v8h b0hi = *(const v8h*)(eb0 + j0 + kbB + 8);
            v16h b0F = __builtin_shufflevector(b0lo, b0hi,
                0,1,2,3,4,5,6,7,8,9,10,11,12,13,14,15);
            v8h b1lo = *(const v8h*)(eb1 + j0 + kbB);
            v8h b1hi = *(const v8h*)(eb1 + j0 + kbB + 8);
            v16h b1F = __builtin_shufflevector(b1lo, b1hi,
                0,1,2,3,4,5,6,7,8,9,10,11,12,13,14,15);
            acc0 = __builtin_amdgcn_wmma_f32_16x16x32_f16(
                false, aF, false, b0F, (short)0, acc0, false, false);
            acc1 = __builtin_amdgcn_wmma_f32_16x16x32_f16(
                false, aF, false, b1F, (short)0, acc1, false, false);
            __syncthreads();      // tileA consumed before next head rebuilds it
        }
    }
    // store: M = r + 8*(lane>=16), N = lane&15; scale 1/H, relu
    int n0 = i0 + ((lane >> 4) << 3);
    #pragma unroll
    for (int r = 0; r < 8; ++r) {
        size_t o = ((size_t)b*NN + n0 + r)*DIMN + (wave<<5) + lc;
        out[o]      = fmaxf(acc0[r]*0.25f, 0.f);
        out[o + 16] = fmaxf(acc1[r]*0.25f, 0.f);
    }
}

// ---------------------------------------------------------------------------
extern "C" void kernel_launch(void* const* d_in, const int* in_sizes, int n_in,
                              void* d_out, int out_size, void* d_ws, size_t ws_size,
                              hipStream_t stream) {
    const float* x    = (const float*)d_in[0];   // B,N,DIM
    const float* adj  = (const float*)d_in[1];   // B,N,N
    const float* W    = (const float*)d_in[2];   // H,DIM,DIM
    const float* bias = (const float*)d_in[3];   // H,DIM
    const float* a    = (const float*)d_in[4];   // H,2*DIM
    float* out = (float*)d_out;

    // workspace layout (~17.3 MB)
    _Float16* embedT = (_Float16*)d_ws;                           // B*H*DIM*N f16 = 16 MB
    float* s_i = (float*)((char*)d_ws + (size_t)BB*HH*DIMN*NN*2); // B*H*N
    float* s_j = s_i + BB*HH*NN;
    float* Mr  = s_j + BB*HH*NN;
    float* Zr  = Mr  + BB*HH*NN;
    float* va1 = Zr  + BB*HH*NN;
    float* va2 = va1 + HH*DIMN;
    float* c1  = va2 + HH*DIMN;
    float* c2  = c1 + 16;

    k_prep<<<1, 1024, 0, stream>>>(W, bias, a, va1, va2, c1, c2);
    k_embed<<<BB*HH*(NN/16), 256, 0, stream>>>(x, W, bias, embedT);
    k_sij<<<BB*HH*NN/256, 256, 0, stream>>>(x, va1, va2, c1, c2, s_i, s_j);
    k_stats<<<BB*NN/8, 256, 0, stream>>>(adj, s_i, s_j, Mr, Zr);
    k_attn_gemm<<<BB*(NN/16), 256, 0, stream>>>(adj, s_i, s_j, Mr, Zr, embedT, out);
}
